// SelfAttention_54915451847336
// MI455X (gfx1250) — matrix operand order repaired and checked by reading, not compiled, not run on hardware
//
#include <hip/hip_runtime.h>
#include <hip/hip_bf16.h>
#include <stdint.h>

typedef __bf16 bf16_t;
typedef __attribute__((ext_vector_type(16))) __bf16 v16bf;
typedef __attribute__((ext_vector_type(8)))  __bf16 v8bf;
typedef __attribute__((ext_vector_type(8)))  float  v8f;
typedef int i32x4 __attribute__((vector_size(16)));

#define WMMA_BF16(A, B, C) \
  __builtin_amdgcn_wmma_f32_16x16x32_bf16(false, (A), false, (B), (short)0, (C), false, false)

static __device__ __forceinline__ v16bf cat16(v8bf lo, v8bf hi) {
  return __builtin_shufflevector(lo, hi, 0,1,2,3,4,5,6,7,8,9,10,11,12,13,14,15);
}

// ---- CDNA5 async global->LDS copy (ASYNCcnt-tracked), 16B per lane ----
static __device__ __forceinline__ void async_cp16(const bf16_t* g, bf16_t* l) {
#if __has_builtin(__builtin_amdgcn_global_load_async_to_lds_b128)
  typedef __attribute__((address_space(1))) i32x4* gp_t;   // global int4*
  typedef __attribute__((address_space(3))) i32x4* lp_t;   // LDS int4*
  __builtin_amdgcn_global_load_async_to_lds_b128(
      (gp_t)(uintptr_t)g, (lp_t)(uintptr_t)l, 0, 0);
#else
  unsigned int loff = (unsigned int)(uintptr_t)l;      // low 32 bits = LDS offset
  unsigned long long ga = (unsigned long long)(uintptr_t)g;
  asm volatile("global_load_async_to_lds_b128 %0, %1, off" :: "v"(loff), "v"(ga) : "memory");
#endif
}

#if __has_builtin(__builtin_amdgcn_s_wait_asynccnt)
#define WAIT_ASYNC(n) __builtin_amdgcn_s_wait_asynccnt(n)
#else
#define WAIT_ASYNC(n) asm volatile("s_wait_asynccnt %0" :: "n"(n) : "memory")
#endif

// ---------------- sizes ----------------
#define NB   4
#define NPIX 4096
#define NC   512
#define ND   64

// ---------------------------------------------------------------------------
// Kernel 1: convert x to bf16; transpose+convert weights to [dout][cin] bf16.
// ---------------------------------------------------------------------------
__global__ void k_prep(const float* __restrict__ x,
                       const float* __restrict__ wf,
                       const float* __restrict__ wg,
                       const float* __restrict__ wh,
                       bf16_t* __restrict__ xb,
                       bf16_t* __restrict__ wft,
                       bf16_t* __restrict__ wgt,
                       bf16_t* __restrict__ wht) {
  const long NX = (long)NB * NPIX * NC;
  const long NF = (long)ND * NC;
  const long NH = (long)NC * NC;
  const long total = NX + 2 * NF + NH;
  for (long i = (long)blockIdx.x * blockDim.x + threadIdx.x; i < total;
       i += (long)gridDim.x * blockDim.x) {
    if (i < NX) {
      xb[i] = (bf16_t)x[i];
    } else if (i < NX + NF) {
      long j = i - NX; long d = j >> 9, c = j & 511;
      wft[j] = (bf16_t)wf[c * ND + d];
    } else if (i < NX + 2 * NF) {
      long j = i - NX - NF; long d = j >> 9, c = j & 511;
      wgt[j] = (bf16_t)wg[c * ND + d];
    } else {
      long j = i - NX - 2 * NF; long d = j >> 9, c = j & 511;
      wht[j] = (bf16_t)wh[c * NC + d];
    }
  }
}

// ---------------------------------------------------------------------------
// Kernel 2: projections via WMMA.  f,g row-major [pix][64]; v transposed
// vT[b][ch][pix] via per-wave LDS transpose. 1280 blocks x 8 waves.
// ---------------------------------------------------------------------------
__global__ __launch_bounds__(256) void k_proj(const bf16_t* __restrict__ xb,
                                              const bf16_t* __restrict__ wft,
                                              const bf16_t* __restrict__ wgt,
                                              const bf16_t* __restrict__ wht,
                                              bf16_t* __restrict__ fb,
                                              bf16_t* __restrict__ gb,
                                              bf16_t* __restrict__ vT) {
  __shared__ bf16_t lds[8 * 1024];
  const int wid  = threadIdx.x >> 5;
  const int lane = threadIdx.x & 31;
  const int l16  = lane & 15;
  const int hf   = lane >> 4;

  const int wgid  = blockIdx.x * 8 + wid;
  const int mtile = wgid / 10;
  const int ngrp  = wgid % 10;

  const bf16_t* wt = (ngrp == 0) ? wft
                   : (ngrp == 1) ? wgt
                   : wht + (size_t)(ngrp - 2) * 64 * NC;

  const int row = mtile * 16 + l16;
  const bf16_t* arow = xb + (size_t)row * NC;

  const v8f zero = {0.f, 0.f, 0.f, 0.f, 0.f, 0.f, 0.f, 0.f};
  v8f acc[4];
#pragma unroll
  for (int t = 0; t < 4; ++t) acc[t] = zero;

#pragma unroll
  for (int ks = 0; ks < 16; ++ks) {
    const int k0 = ks * 32;
    v8bf alo = *(const v8bf*)(arow + k0 + 16 * hf);
    v8bf ahi = *(const v8bf*)(arow + k0 + 16 * hf + 8);
    v16bf A = cat16(alo, ahi);
#pragma unroll
    for (int t = 0; t < 4; ++t) {
      const bf16_t* bp = wt + (size_t)(t * 16 + l16) * NC + k0 + 16 * hf;
      v16bf Bm = *(const v16bf*)bp;
      acc[t] = WMMA_BF16(A, Bm, acc[t]);
    }
  }

  if (ngrp < 2) {
    bf16_t* outp = (ngrp == 0) ? fb : gb;
#pragma unroll
    for (int t = 0; t < 4; ++t)
#pragma unroll
      for (int r = 0; r < 8; ++r) {
        int gr = mtile * 16 + r + 8 * hf;
        outp[(size_t)gr * ND + t * 16 + l16] = (bf16_t)acc[t][r];
      }
  } else {
    bf16_t* w = lds + wid * 1024;   // per-wave 64 cols x 16 pix transpose tile
#pragma unroll
    for (int t = 0; t < 4; ++t)
#pragma unroll
      for (int r = 0; r < 8; ++r)
        w[(t * 16 + l16) * 16 + r + 8 * hf] = (bf16_t)acc[t][r];
    const int pixBase = mtile * 16;
    const int b = pixBase >> 12;
    const int pixL = pixBase & (NPIX - 1);
#pragma unroll
    for (int cc = 0; cc < 2; ++cc) {
      int colL = lane + cc * 32;
      v16bf vv = *(const v16bf*)(w + colL * 16);
      int chG = (ngrp - 2) * 64 + colL;
      *(v16bf*)(vT + ((size_t)(b * NC + chG) * NPIX + pixL)) = vv;
    }
  }
}

// ---------------------------------------------------------------------------
// Kernel 3: fused flash attention, g-block double-buffered via async->LDS.
// 512 blocks x 8 waves. Wave (mi,ci): 16 rows x 128 channels.
// ---------------------------------------------------------------------------
__global__ __launch_bounds__(256) void k_attn(const bf16_t* __restrict__ fbuf,
                                              const bf16_t* __restrict__ gbuf,
                                              const bf16_t* __restrict__ vT,
                                              const float* __restrict__ x,
                                              const float* __restrict__ gamma,
                                              float* __restrict__ out) {
  __shared__ bf16_t gsh[2][64 * 64];   // double-buffered g block (2 x 8KB)
  __shared__ bf16_t psh[8][1024];      // per-wave 16x64 beta tiles
  const int tid  = threadIdx.x;
  const int wid  = tid >> 5;
  const int lane = tid & 31;
  const int l16  = lane & 15;
  const int hf   = lane >> 4;
  const int mi   = wid >> 2;
  const int ci   = wid & 3;
  const int b    = blockIdx.x >> 7;
  const int mb   = ((blockIdx.x & 127) * 32) + mi * 16;
  const int cb   = ci * 128;

  // A-operand: 16 rows of f, register-resident for the whole loop
  const bf16_t* frow = fbuf + (size_t)(b * NPIX + mb + l16) * ND;
  v16bf fA[2];
#pragma unroll
  for (int kh = 0; kh < 2; ++kh) {
    v8bf lo = *(const v8bf*)(frow + kh * 32 + 16 * hf);
    v8bf hi = *(const v8bf*)(frow + kh * 32 + 16 * hf + 8);
    fA[kh] = cat16(lo, hi);
  }

  const v8f zero = {0.f, 0.f, 0.f, 0.f, 0.f, 0.f, 0.f, 0.f};
  v8f accO[8];
  float mrow[8], lrow[8];
#pragma unroll
  for (int t = 0; t < 8; ++t) { accO[t] = zero; mrow[t] = -3.0e38f; lrow[t] = 0.f; }

  bf16_t* P = psh[wid];
  const bf16_t* gbase = gbuf + (size_t)b * NPIX * ND;  // g block rows contiguous
  const bf16_t* vbase = vT + (size_t)b * NC * NPIX;

  // prologue: stage g block for jb=0 (each thread: 2 async b128 = 32B)
#pragma unroll
  for (int q = 0; q < 2; ++q) {
    int c = tid + q * 256;                      // 512 chunks x 16B = 8KB
    async_cp16(gbase + c * 8, &gsh[0][c * 8]);
  }

  for (int it = 0; it < NPIX / 64; ++it) {
    const int jb = it * 64;
    const int cur = it & 1;
    if (it + 1 < NPIX / 64) {
#pragma unroll
      for (int q = 0; q < 2; ++q) {
        int c = tid + q * 256;
        async_cp16(gbase + (size_t)(jb + 64) * ND + c * 8, &gsh[cur ^ 1][c * 8]);
      }
      WAIT_ASYNC(2);     // previous batch (current buffer) complete
    } else {
      WAIT_ASYNC(0);
    }
    __syncthreads();     // all waves' async copies for gsh[cur] visible

    const bf16_t* gl = gsh[cur];

    // ---- S = f @ g^T for a 16x64 block (B-operand from LDS) ----
    v8f s[4];
#pragma unroll
    for (int t = 0; t < 4; ++t) s[t] = zero;
#pragma unroll
    for (int t = 0; t < 4; ++t) {
      const bf16_t* gr = gl + (t * 16 + l16) * ND;
#pragma unroll
      for (int kh = 0; kh < 2; ++kh) {
        v16bf Bm = *(const v16bf*)(gr + kh * 32 + 16 * hf);
        s[t] = WMMA_BF16(fA[kh], Bm, s[t]);
      }
    }

    // ---- online softmax (row M = r + 8*hf lives across 16 lanes) ----
#pragma unroll
    for (int r = 0; r < 8; ++r) {
      float mx = fmaxf(fmaxf(s[0][r], s[1][r]), fmaxf(s[2][r], s[3][r]));
#pragma unroll
      for (int off = 1; off < 16; off <<= 1)
        mx = fmaxf(mx, __shfl_xor(mx, off, 32));
      float mnew  = fmaxf(mrow[r], mx);
      float alpha = __expf(mrow[r] - mnew);
      mrow[r] = mnew;
      float rs = 0.f;
#pragma unroll
      for (int t = 0; t < 4; ++t) {
        float p = __expf(s[t][r] - mnew);
        s[t][r] = p;
        rs += p;
      }
#pragma unroll
      for (int off = 1; off < 16; off <<= 1) rs += __shfl_xor(rs, off, 32);
      lrow[r] = lrow[r] * alpha + rs;
#pragma unroll
      for (int ct = 0; ct < 8; ++ct) accO[ct][r] *= alpha;
#pragma unroll
      for (int t = 0; t < 4; ++t)
        P[(r + 8 * hf) * 64 + t * 16 + l16] = (bf16_t)s[t][r];
    }

    // ---- out += beta @ v  (B-operand from vT: contiguous along j) ----
#pragma unroll
    for (int kh = 0; kh < 2; ++kh) {
      v8bf lo = *(const v8bf*)(P + l16 * 64 + kh * 32 + 16 * hf);
      v8bf hi = *(const v8bf*)(P + l16 * 64 + kh * 32 + 16 * hf + 8);
      v16bf pA = cat16(lo, hi);
#pragma unroll
      for (int ct = 0; ct < 8; ++ct) {
        const bf16_t* vr =
            vbase + (size_t)(cb + ct * 16 + l16) * NPIX + jb + kh * 32 + 16 * hf;
        v16bf Bm = *(const v16bf*)vr;
        accO[ct] = WMMA_BF16(pA, Bm, accO[ct]);
      }
    }

    // prefetch next j-block of vT rows this wave will need
    if (it + 1 < NPIX / 64) {
#pragma unroll
      for (int ct = 0; ct < 8; ++ct)
        __builtin_prefetch(vbase + (size_t)(cb + ct * 16 + l16) * NPIX + jb + 64, 0, 3);
    }

    __syncthreads();     // everyone done reading gsh[cur] before it's reused
  }

  // ---- epilogue: out = gamma * (acc / l) + x ----
  const float gm = gamma[0];
#pragma unroll
  for (int ct = 0; ct < 8; ++ct)
#pragma unroll
    for (int r = 0; r < 8; ++r) {
      int pix = mb + r + 8 * hf;
      int ch  = cb + ct * 16 + l16;
      size_t idx = ((size_t)(b * NPIX + pix)) * NC + ch;
      out[idx] = gm * (accO[ct][r] / lrow[r]) + x[idx];
    }
}

// ---------------------------------------------------------------------------
extern "C" void kernel_launch(void* const* d_in, const int* in_sizes, int n_in,
                              void* d_out, int out_size, void* d_ws, size_t ws_size,
                              hipStream_t stream) {
  const float* x  = (const float*)d_in[0];
  const float* wf = (const float*)d_in[1];
  const float* wg = (const float*)d_in[2];
  const float* wh = (const float*)d_in[3];
  const float* gm = (const float*)d_in[4];
  float* out = (float*)d_out;

  char* ws = (char*)d_ws;
  bf16_t* xb  = (bf16_t*)(ws + 0);
  bf16_t* fb  = (bf16_t*)(ws + 16777216);
  bf16_t* gb  = (bf16_t*)(ws + 18874368);
  bf16_t* vT  = (bf16_t*)(ws + 20971520);
  bf16_t* wft = (bf16_t*)(ws + 37748736);
  bf16_t* wgt = (bf16_t*)(ws + 37814272);
  bf16_t* wht = (bf16_t*)(ws + 37879808);

  k_prep<<<4096, 256, 0, stream>>>(x, wf, wg, wh, xb, wft, wgt, wht);
  k_proj<<<1280, 256, 0, stream>>>(xb, wft, wgt, wht, fb, gb, vT);
  k_attn<<<512, 256, 0, stream>>>(fb, gb, vT, x, gm, out);
}
